// MotionEnergyModule_39470749450511
// MI455X (gfx1250) — compile-verified, matrix operand order checked by reading, not run on hardware
//
#include <hip/hip_runtime.h>

// ---------------------------------------------------------------------------
// MotionEnergy for MI455X (gfx1250): bf16 WMMA GEMM + temporal-conv epilogue
//   stage 1: fp32 -> bf16 convert (x, [w_s_sin ; w_s_cos])
//   stage 2: C(1920 x 5120) = Xb(1920 x 9216) @ Wb^T  via v_wmma_f32_16x16x32_bf16
//            double-buffered LDS; async global->LDS copies (ASYNCcnt-tracked)
//   stage 3: depthwise 16-tap conv along T + log-energy
// Workspace: Xb bf16 35,389,440 B | Wb bf16 94,371,840 B | C f32 39,321,600 B
// ---------------------------------------------------------------------------

typedef __attribute__((ext_vector_type(16))) __bf16 v16bf;
typedef __attribute__((ext_vector_type(8)))  float  v8f;

// exact operand types for the async-LDS builtin (from hipcc diagnostic):
//   param0: int __vector(4) __device__ *   (global, AS1)
//   param1: LDS-side pointer               (AS3)
typedef int v4i_t __attribute__((vector_size(16)));
typedef __attribute__((address_space(1))) v4i_t* gv4i_p;
typedef __attribute__((address_space(3))) v4i_t* lv4i_p;

#define NFILT 2560
#define TWIN  16
#define BATCH 8
#define TLEN  240
#define MDIM  (BATCH * TLEN)   // 1920
#define KDIM  (96 * 96)        // 9216
#define NDIM  (2 * NFILT)      // 5120
#define PADL  8

#define M_BLK  128
#define N_BLK  128
#define K_TILE 64
#define LDS_K  (K_TILE + 8)            // rows stay 16B aligned, banks skewed
#define NTILES (KDIM / K_TILE)         // 144

#if __has_builtin(__builtin_amdgcn_global_load_async_to_lds_b128) && \
    __has_builtin(__builtin_amdgcn_s_wait_asynccnt)
#define USE_ASYNC_LDS 1
#else
#define USE_ASYNC_LDS 0
#endif

static __device__ __forceinline__ unsigned short f2bf(float f) {
  union { float f; unsigned u; } c; c.f = f;
  unsigned r = c.u + 0x7FFFu + ((c.u >> 16) & 1u);   // round-to-nearest-even
  return (unsigned short)(r >> 16);
}

// ---------------- stage 1: fp32 -> bf16, 4 elems / thread --------------------
__global__ void cvt_bf16_kernel(const float* __restrict__ in,
                                unsigned short* __restrict__ out, int n4) {
  int i = blockIdx.x * blockDim.x + threadIdx.x;
  if (i < n4) {
    float4 v = reinterpret_cast<const float4*>(in)[i];
    ushort4 o;
    o.x = f2bf(v.x); o.y = f2bf(v.y); o.z = f2bf(v.z); o.w = f2bf(v.w);
    reinterpret_cast<ushort4*>(out)[i] = o;
  }
}

// ---------------- stage 2: bf16 WMMA GEMM -----------------------------------
// A: MDIM x KDIM (row major, bf16)   Bw: NDIM x KDIM (row major, bf16)
// C: MDIM x NDIM (f32) = A @ Bw^T
__global__ __launch_bounds__(256) void gemm_bf16_wmma(
    const unsigned short* __restrict__ A,
    const unsigned short* __restrict__ Bw,
    float* __restrict__ C) {
  __shared__ unsigned short Xs[2][M_BLK][LDS_K];
  __shared__ unsigned short Ws[2][N_BLK][LDS_K];

  const int tid  = threadIdx.x;
  const int lane = tid & 31;
  const int half = (lane >> 4) & 1;   // lane half selects K sub-range / M+8
  const int r    = lane & 15;
  const int wave = tid >> 5;          // 8 waves
  const int wm   = (wave & 1) * 64;   // wave M offset in block tile
  const int wn   = (wave >> 1) * 32;  // wave N offset in block tile

  const int n0 = blockIdx.x * N_BLK;
  const int m0 = blockIdx.y * M_BLK;

  // staging map: 256 threads * 4 rounds * 16B = 128 rows x 64 cols bf16
  const int tr = tid >> 3;            // 0..31 (row within round)
  const int tc = (tid & 7) * 8;       // element col 0..56 (16B chunks)

  v8f zero = {};
  v8f acc[4][2];
#pragma unroll
  for (int i = 0; i < 4; ++i)
#pragma unroll
    for (int j = 0; j < 2; ++j) acc[i][j] = zero;

#if USE_ASYNC_LDS
  // ---- async global->LDS double-buffered pipeline (ASYNCcnt-tracked) ----
  {
    // prologue: issue tile 0 into buffer 0 (8 async b128 ops per thread/tile)
#pragma unroll
    for (int p = 0; p < 4; ++p) {
      const int row = tr + p * 32;
      __builtin_amdgcn_global_load_async_to_lds_b128(
          (gv4i_p)&A[(size_t)(m0 + row) * KDIM + tc],
          (lv4i_p)&Xs[0][row][tc], 0, 0);
      __builtin_amdgcn_global_load_async_to_lds_b128(
          (gv4i_p)&Bw[(size_t)(n0 + row) * KDIM + tc],
          (lv4i_p)&Ws[0][row][tc], 0, 0);
    }
  }
  for (int it = 0; it < NTILES; ++it) {
    const int cur = it & 1;
    if (it + 1 < NTILES) {
      const int k0 = (it + 1) * K_TILE;
#pragma unroll
      for (int p = 0; p < 4; ++p) {
        const int row = tr + p * 32;
        __builtin_amdgcn_global_load_async_to_lds_b128(
            (gv4i_p)&A[(size_t)(m0 + row) * KDIM + k0 + tc],
            (lv4i_p)&Xs[cur ^ 1][row][tc], 0, 0);
        __builtin_amdgcn_global_load_async_to_lds_b128(
            (gv4i_p)&Bw[(size_t)(n0 + row) * KDIM + k0 + tc],
            (lv4i_p)&Ws[cur ^ 1][row][tc], 0, 0);
      }
      __builtin_amdgcn_s_wait_asynccnt(8);  // in-order: tile `it` has landed
    } else {
      __builtin_amdgcn_s_wait_asynccnt(0);
    }
    __syncthreads();

#pragma unroll
    for (int kb = 0; kb < K_TILE; kb += 32) {
      const int klo = kb + half * 8;
      v16bf a[4], b[2];
#pragma unroll
      for (int i = 0; i < 4; ++i) {
        uint4* ap = reinterpret_cast<uint4*>(&a[i]);
        ap[0] = *reinterpret_cast<const uint4*>(&Xs[cur][wm + i * 16 + r][klo]);
        ap[1] = *reinterpret_cast<const uint4*>(&Xs[cur][wm + i * 16 + r][klo + 16]);
      }
#pragma unroll
      for (int j = 0; j < 2; ++j) {
        uint4* bp = reinterpret_cast<uint4*>(&b[j]);
        bp[0] = *reinterpret_cast<const uint4*>(&Ws[cur][wn + j * 16 + r][klo]);
        bp[1] = *reinterpret_cast<const uint4*>(&Ws[cur][wn + j * 16 + r][klo + 16]);
      }
#pragma unroll
      for (int i = 0; i < 4; ++i)
#pragma unroll
        for (int j = 0; j < 2; ++j)
          acc[i][j] = __builtin_amdgcn_wmma_f32_16x16x32_bf16(
              false, a[i], false, b[j], (short)0, acc[i][j], false, false);
    }
    __syncthreads();
  }
#else
  // ---- fallback: register double-buffer (global loads overlap compute) ----
  uint4 xr[4], wr[4];
#pragma unroll
  for (int p = 0; p < 4; ++p) {
    const int row = tr + p * 32;
    xr[p] = *reinterpret_cast<const uint4*>(&A[(size_t)(m0 + row) * KDIM + tc]);
    wr[p] = *reinterpret_cast<const uint4*>(&Bw[(size_t)(n0 + row) * KDIM + tc]);
  }
  for (int it = 0; it < NTILES; ++it) {
    const int cur = it & 1;
#pragma unroll
    for (int p = 0; p < 4; ++p) {
      const int row = tr + p * 32;
      *reinterpret_cast<uint4*>(&Xs[cur][row][tc]) = xr[p];
      *reinterpret_cast<uint4*>(&Ws[cur][row][tc]) = wr[p];
    }
    if (it + 1 < NTILES) {
      const int k0 = (it + 1) * K_TILE;
#pragma unroll
      for (int p = 0; p < 4; ++p) {
        const int row = tr + p * 32;
        xr[p] = *reinterpret_cast<const uint4*>(&A[(size_t)(m0 + row) * KDIM + k0 + tc]);
        wr[p] = *reinterpret_cast<const uint4*>(&Bw[(size_t)(n0 + row) * KDIM + k0 + tc]);
      }
    }
    __syncthreads();

#pragma unroll
    for (int kb = 0; kb < K_TILE; kb += 32) {
      const int klo = kb + half * 8;
      v16bf a[4], b[2];
#pragma unroll
      for (int i = 0; i < 4; ++i) {
        uint4* ap = reinterpret_cast<uint4*>(&a[i]);
        ap[0] = *reinterpret_cast<const uint4*>(&Xs[cur][wm + i * 16 + r][klo]);
        ap[1] = *reinterpret_cast<const uint4*>(&Xs[cur][wm + i * 16 + r][klo + 16]);
      }
#pragma unroll
      for (int j = 0; j < 2; ++j) {
        uint4* bp = reinterpret_cast<uint4*>(&b[j]);
        bp[0] = *reinterpret_cast<const uint4*>(&Ws[cur][wn + j * 16 + r][klo]);
        bp[1] = *reinterpret_cast<const uint4*>(&Ws[cur][wn + j * 16 + r][klo + 16]);
      }
#pragma unroll
      for (int i = 0; i < 4; ++i)
#pragma unroll
        for (int j = 0; j < 2; ++j)
          acc[i][j] = __builtin_amdgcn_wmma_f32_16x16x32_bf16(
              false, a[i], false, b[j], (short)0, acc[i][j], false, false);
    }
    __syncthreads();
  }
#endif

  // epilogue: C/D 16x16 f32 layout -> VGPR e holds M = e + half*8, N = r
#pragma unroll
  for (int i = 0; i < 4; ++i)
#pragma unroll
    for (int j = 0; j < 2; ++j) {
      const int gn = n0 + wn + j * 16 + r;
#pragma unroll
      for (int e = 0; e < 8; ++e) {
        const int gm = m0 + wm + i * 16 + half * 8 + e;
        C[(size_t)gm * NDIM + gn] = acc[i][j][e];
      }
    }
}

// ---------------- stage 3: temporal depthwise conv + log energy -------------
// G: MDIM x NDIM; cols [0,NFILT) = g_sin, [NFILT,2*NFILT) = g_cos
__global__ __launch_bounds__(256) void energy_kernel(
    const float* __restrict__ G,
    const float* __restrict__ wts,   // (NF, TW) time-flipped sin taps
    const float* __restrict__ wtc,   // (NF, TW) time-flipped cos taps
    float* __restrict__ out) {       // (B, T, NF)
  __shared__ float gs[TLEN], gc[TLEN], wsl[TWIN], wcl[TWIN];
  const int bf = blockIdx.x;
  const int b  = bf / NFILT;
  const int f  = bf % NFILT;
  const int t  = threadIdx.x;

  if (t < TWIN) { wsl[t] = wts[f * TWIN + t]; wcl[t] = wtc[f * TWIN + t]; }
  if (t < TLEN) {
    const size_t base = (size_t)(b * TLEN + t) * NDIM;
    gs[t] = G[base + f];
    gc[t] = G[base + NFILT + f];
  }
  __syncthreads();

  if (t < TLEN) {
    float rs = 0.f, rc = 0.f;
#pragma unroll
    for (int k = 0; k < TWIN; ++k) {
      const int tt = t - PADL + k;
      const bool ok = (tt >= 0) && (tt < TLEN);
      const float s = ok ? gs[tt] : 0.f;
      const float c = ok ? gc[tt] : 0.f;
      rs += s * wcl[k] + c * wsl[k];   // term1 + term2
      rc += c * wcl[k] - s * wsl[k];   // term3 - term4
    }
    out[(size_t)(b * TLEN + t) * NFILT + f] =
        logf(sqrtf(rs * rs + rc * rc) + 1e-5f);
  }
}

// ---------------------------------------------------------------------------
extern "C" void kernel_launch(void* const* d_in, const int* in_sizes, int n_in,
                              void* d_out, int out_size, void* d_ws, size_t ws_size,
                              hipStream_t stream) {
  (void)in_sizes; (void)n_in; (void)out_size; (void)ws_size;

  const float* x    = (const float*)d_in[0];  // (B,T,H,W)
  const float* wssi = (const float*)d_in[1];  // (NF,H,W)
  const float* wsco = (const float*)d_in[2];  // (NF,H,W)
  const float* wtsi = (const float*)d_in[3];  // (NF,TW)
  const float* wtco = (const float*)d_in[4];  // (NF,TW)
  float* out = (float*)d_out;

  char* ws = (char*)d_ws;
  const size_t XB_BYTES = (size_t)MDIM * KDIM * 2;        // 35,389,440
  const size_t WB_BYTES = (size_t)NDIM * KDIM * 2;        // 94,371,840
  unsigned short* Xb = (unsigned short*)ws;
  unsigned short* Wb = (unsigned short*)(ws + XB_BYTES);
  float*          Cg = (float*)(ws + XB_BYTES + WB_BYTES);

  // stage 1: converts
  {
    const int nx4 = (MDIM * KDIM) / 4;                    // 4,423,680
    cvt_bf16_kernel<<<(nx4 + 255) / 256, 256, 0, stream>>>(x, Xb, nx4);
    const int nw4 = (NFILT * KDIM) / 4;                   // 5,898,240
    cvt_bf16_kernel<<<(nw4 + 255) / 256, 256, 0, stream>>>(wssi, Wb, nw4);
    cvt_bf16_kernel<<<(nw4 + 255) / 256, 256, 0, stream>>>(
        wsco, Wb + (size_t)NFILT * KDIM, nw4);
  }

  // stage 2: WMMA GEMM, grid = (NDIM/128, MDIM/128) = (40, 15)
  gemm_bf16_wmma<<<dim3(NDIM / N_BLK, MDIM / M_BLK), 256, 0, stream>>>(Xb, Wb, Cg);

  // stage 3: energy, one block per (b, f)
  energy_kernel<<<BATCH * NFILT, 256, 0, stream>>>(Cg, wtsi, wtco, out);
}
